// GraphAttentionLayer_78348793413869
// MI455X (gfx1250) — matrix operand order repaired and checked by reading, not compiled, not run on hardware
//
#include <hip/hip_runtime.h>
#include <hip/hip_bf16.h>
#include <stdint.h>

typedef __attribute__((ext_vector_type(16))) _Float16 v16h;
typedef __attribute__((ext_vector_type(8)))  _Float16 v8h;
typedef __attribute__((ext_vector_type(8)))  float    v8f;

#define UNITS  128
#define HEADS  4
#define IN_DIM 256
#define NV     1024
#define BV     8
#define NEGV   (-1e9f)
#define MSTEP  64
#define STEPS  (NV / MSTEP)

__device__ __forceinline__ float lrelu(float v) { return v > 0.f ? v : 0.2f * v; }

__device__ __forceinline__ void async_lds_b128(uint32_t lds_off, uint64_t gaddr) {
    // GLOBAL_LOAD_ASYNC_TO_LDS_B128: per-lane global->LDS DMA, tracked by ASYNCcnt
    asm volatile("global_load_async_to_lds_b128 %0, %1, off" :: "v"(lds_off), "v"(gaddr) : "memory");
}
__device__ __forceinline__ void wait_async0() {
    asm volatile("s_wait_asynccnt 0x0" ::: "memory");
}

// ---------------- kernel 0a: W [H,F,U] f32 -> WT [H,U,F] f16 (transpose: B-frags contiguous)
__global__ void k_prep_w(const float* __restrict__ W, _Float16* __restrict__ WT) {
    int idx = blockIdx.x * blockDim.x + threadIdx.x;           // over H*F*U
    int h   = idx / (IN_DIM * UNITS);
    int rem = idx % (IN_DIM * UNITS);
    int f   = rem / UNITS;
    int u   = rem % UNITS;
    WT[((size_t)h * UNITS + u) * IN_DIM + f] = (_Float16)W[idx];
}

// ---------------- kernel 0b: x f32 -> f16 (so the GEMM can async-DMA it straight into LDS)
__global__ void k_prep_x(const float* __restrict__ x, _Float16* __restrict__ xh) {
    size_t i = (size_t)blockIdx.x * blockDim.x + threadIdx.x;  // over B*N*F
    xh[i] = (_Float16)x[i];
}

// ---------------- kernel 1: h = x @ W (per head), writes hT [B,H,U,N] f16
// 256 thr = 8 waves; each wave one 16x16 tile; 8 waves cover U=128. x-strip staged by async DMA.
__global__ void k_gemm_h(const _Float16* __restrict__ xh, const _Float16* __restrict__ WT,
                         _Float16* __restrict__ hT) {
    __shared__ __align__(16) _Float16 xs[16][IN_DIM + 8];
    const int m0   = blockIdx.x * 16;          // flat row tile over B*N
    const int head = blockIdx.y;
    const int tid  = threadIdx.x;

    // 16 rows x 256 halves = 512 x 16B chunks; 2 chunks per thread via async-to-LDS
#pragma unroll
    for (int c = tid; c < 512; c += 256) {
        int r    = c >> 5;
        int col8 = (c & 31) * 8;
        uint64_t g = (uint64_t)(uintptr_t)(xh + (size_t)(m0 + r) * IN_DIM + col8);
        uint32_t l = (uint32_t)(uintptr_t)&xs[r][col8];
        async_lds_b128(l, g);
    }
    wait_async0();
    __syncthreads();

    const int lane = tid & 31;
    const int wave = tid >> 5;
    const int hi   = lane >> 4;
    const int row  = lane & 15;
    const int uCol = wave * 16 + (lane & 15);

    const _Float16* wrow = WT + ((size_t)head * UNITS + uCol) * IN_DIM;

    v8f c = {};
#pragma unroll
    for (int k0 = 0; k0 < IN_DIM; k0 += 32) {
        // A-frag: halves i -> K = k0 + hi*16 + i
        v8h alo = *(const v8h*)&xs[row][k0 + hi * 16];
        v8h ahi = *(const v8h*)&xs[row][k0 + hi * 16 + 8];
        v16h a  = __builtin_shufflevector(alo, ahi, 0,1,2,3,4,5,6,7,8,9,10,11,12,13,14,15);
        // B-frag: lane col uCol, halves i -> K = k0 + hi*16 + i (contiguous in WT)
        v16h b  = *(const v16h*)(wrow + k0 + hi * 16);
        c = __builtin_amdgcn_wmma_f32_16x16x32_f16(false, a, false, b, (short)0, c, false, false);
    }

    // C layout: VGPR r -> M = r + 8*hi, N = lane&15
#pragma unroll
    for (int vr = 0; vr < 8; ++vr) {
        int rflat = m0 + vr + 8 * hi;
        int b     = rflat >> 10;
        int n     = rflat & (NV - 1);
        hT[(((size_t)b * HEADS + head) * UNITS + uCol) * NV + n] = (_Float16)c[vr];
    }
}

// ---------------- kernel 2: src/dst = h . a_src / a_dst (one block per (b,h))
__global__ void k_srcdst(const _Float16* __restrict__ hT, const float* __restrict__ a_src,
                         const float* __restrict__ a_dst, float* __restrict__ srcO,
                         float* __restrict__ dstO) {
    int bh = blockIdx.x;                 // b*H + h
    int h  = bh % HEADS;
    for (int n = threadIdx.x; n < NV; n += blockDim.x) {
        float s = 0.f, d = 0.f;
        const _Float16* base = hT + (size_t)bh * UNITS * NV + n;
        for (int u = 0; u < UNITS; ++u) {
            float v = (float)base[(size_t)u * NV];
            s = fmaf(v, a_src[h * UNITS + u], s);
            d = fmaf(v, a_dst[h * UNITS + u], d);
        }
        srcO[bh * NV + n] = s;
        dstO[bh * NV + n] = d;
    }
}

// ---------------- kernel 3: per-row online softmax stats (wave per (b,h,n))
__global__ void k_rowstats(const float* __restrict__ adj, const float* __restrict__ srcV,
                           const float* __restrict__ dstV, float* __restrict__ rowmax,
                           float* __restrict__ rowsum) {
    int wg   = blockIdx.x * (blockDim.x >> 5) + (threadIdx.x >> 5);  // over B*H*N
    int lane = threadIdx.x & 31;
    int bh   = wg >> 10;
    int n    = wg & (NV - 1);
    float sv = srcV[bh * NV + n];
    float M = -3.0e38f, S = 0.f;
    for (int m = lane; m < NV; m += 32) {
        bool mask = (adj[(size_t)n * NV + m] > 0.f) || (m == n);
        float s = mask ? lrelu(sv + dstV[bh * NV + m]) : NEGV;
        float nM = fmaxf(M, s);
        S = S * __expf(M - nM) + __expf(s - nM);
        M = nM;
    }
    for (int off = 16; off > 0; off >>= 1) {
        float oM = __shfl_xor(M, off, 32);
        float oS = __shfl_xor(S, off, 32);
        float nM = fmaxf(M, oM);
        S = S * __expf(M - nM) + oS * __expf(oM - nM);
        M = nM;
    }
    if (lane == 0) { rowmax[wg] = M; rowsum[wg] = S; }
}

// ---------------- kernel 4: out = softmax(scores) @ h, fused, async-double-buffered adj tiles
__global__ void k_attn(const float* __restrict__ adj, const _Float16* __restrict__ hT,
                       const float* __restrict__ srcV, const float* __restrict__ dstV,
                       const float* __restrict__ rowmax, const float* __restrict__ rowsum,
                       float* __restrict__ out) {
    __shared__ __align__(16) _Float16 P[2][16][MSTEP + 8];     // 16 x 64 probability tile, x2
    __shared__ __align__(16) float    adjS[2][16][MSTEP + 4];  // 16 x 64 adjacency tile, x2
    const int n0   = blockIdx.x * 16;
    const int head = blockIdx.y;
    const int b    = blockIdx.z;
    const int bh   = b * HEADS + head;
    const int tid  = threadIdx.x;
    const int lane = tid & 31;
    const int wave = tid >> 5;
    const int hi   = lane >> 4;
    const int row  = lane & 15;
    const int uCol = wave * 16 + (lane & 15);

    // async chunk mapping: 16 rows x 64 f32 = 256 x 16B chunks; exactly one per thread
    const int crow = tid >> 4;            // 0..15
    const int ccol = (tid & 15) * 4;      // f32 column offset within tile
    const uint64_t gAdjRow = (uint64_t)(uintptr_t)(adj + (size_t)(n0 + crow) * NV + ccol);

    // P-tile mapping: fixed m-column per thread, 4 rows (stride 4)
    const int ml  = tid & 63;
    const int nlb = tid >> 6;             // 0..3
    float sv[4], mx[4], is[4];
#pragma unroll
    for (int j = 0; j < 4; ++j) {
        int n = n0 + nlb + 4 * j;
        sv[j] = srcV[bh * NV + n];
        mx[j] = rowmax[bh * NV + n];
        is[j] = 1.f / rowsum[bh * NV + n];
    }

    const _Float16* hrow = hT + ((size_t)bh * UNITS + uCol) * NV;

    // prologue: DMA adj tile 0 into buffer 0
    async_lds_b128((uint32_t)(uintptr_t)&adjS[0][crow][ccol], gAdjRow);

    v8f c = {};
    for (int it = 0; it < STEPS; ++it) {
        const int buf = it & 1;
        const int m0  = it * MSTEP;

        wait_async0();                      // my chunks of adj tile `it` are in LDS
        __syncthreads();                    // everyone's chunks visible; P[buf] consumers done

        if (it + 1 < STEPS)                 // DMA tile it+1; overlaps exp + WMMA below
            async_lds_b128((uint32_t)(uintptr_t)&adjS[buf ^ 1][crow][ccol],
                           gAdjRow + (uint64_t)(m0 + MSTEP) * 4u);

        // P[n,m] = exp(lrelu(src+dst)-max)/sum, 0 where masked
        const float dv = dstV[bh * NV + m0 + ml];
#pragma unroll
        for (int j = 0; j < 4; ++j) {
            int nl = nlb + 4 * j;
            bool mask = (adjS[buf][nl][ml] > 0.f) || (m0 + ml == n0 + nl);
            float p = mask ? __expf(lrelu(sv[j] + dv) - mx[j]) * is[j] : 0.f;
            P[buf][nl][ml] = (_Float16)p;
        }
        __syncthreads();                    // P[buf] published

        __builtin_prefetch(hrow + m0 + 2 * MSTEP, 0, 1);   // global_prefetch_b8, h strip two steps ahead

        // two 16x16x32 WMMAs cover the 64-wide m step
        v8h alo0 = *(const v8h*)&P[buf][row][hi * 16];
        v8h ahi0 = *(const v8h*)&P[buf][row][hi * 16 + 8];
        v16h a0  = __builtin_shufflevector(alo0, ahi0, 0,1,2,3,4,5,6,7,8,9,10,11,12,13,14,15);
        v16h b0  = *(const v16h*)(hrow + m0 + hi * 16);
        c = __builtin_amdgcn_wmma_f32_16x16x32_f16(false, a0, false, b0, (short)0, c, false, false);

        v8h alo1 = *(const v8h*)&P[buf][row][32 + hi * 16];
        v8h ahi1 = *(const v8h*)&P[buf][row][40 + hi * 16];
        v16h a1  = __builtin_shufflevector(alo1, ahi1, 0,1,2,3,4,5,6,7,8,9,10,11,12,13,14,15);
        v16h b1  = *(const v16h*)(hrow + m0 + 32 + hi * 16);
        c = __builtin_amdgcn_wmma_f32_16x16x32_f16(false, a1, false, b1, (short)0, c, false, false);
    }

#pragma unroll
    for (int vr = 0; vr < 8; ++vr) {
        int n = n0 + vr + 8 * hi;
        out[((size_t)b * NV + n) * (HEADS * UNITS) + head * UNITS + uCol] = c[vr];
    }
}

extern "C" void kernel_launch(void* const* d_in, const int* in_sizes, int n_in,
                              void* d_out, int out_size, void* d_ws, size_t ws_size,
                              hipStream_t stream) {
    (void)in_sizes; (void)n_in; (void)out_size; (void)ws_size;
    const float* x     = (const float*)d_in[0];   // [B,N,F]
    const float* adj   = (const float*)d_in[1];   // [N,N]
    const float* W     = (const float*)d_in[2];   // [H,F,U]
    const float* a_src = (const float*)d_in[3];   // [H,U,1]
    const float* a_dst = (const float*)d_in[4];   // [H,U,1]
    float* out = (float*)d_out;                   // [B,N,H*U]

    // workspace carve-up (~12.9 MB)
    _Float16* WT = (_Float16*)d_ws;                                   // H*U*F f16
    _Float16* hT = WT + (size_t)HEADS * UNITS * IN_DIM;               // B*H*U*N f16
    float* srcV  = (float*)(hT + (size_t)BV * HEADS * UNITS * NV);    // B*H*N f32
    float* dstV  = srcV + (size_t)BV * HEADS * NV;
    float* rmax  = dstV + (size_t)BV * HEADS * NV;
    float* rsum  = rmax + (size_t)BV * HEADS * NV;
    _Float16* xh = (_Float16*)(rsum + (size_t)BV * HEADS * NV);       // B*N*F f16

    k_prep_w  <<<(HEADS * IN_DIM * UNITS) / 256, 256, 0, stream>>>(W, WT);
    k_prep_x  <<<(BV * NV * IN_DIM) / 256, 256, 0, stream>>>(x, xh);
    k_gemm_h  <<<dim3((BV * NV) / 16, HEADS), 256, 0, stream>>>(xh, WT, hT);
    k_srcdst  <<<BV * HEADS, 256, 0, stream>>>(hT, a_src, a_dst, srcV, dstV);
    k_rowstats<<<(BV * HEADS * NV) / 8, 256, 0, stream>>>(adj, srcV, dstV, rmax, rsum);
    k_attn    <<<dim3(NV / 16, HEADS, BV), 256, 0, stream>>>(adj, hT, srcV, dstV, rmax, rsum, out);
}
